// AttentionBlock_90975997264646
// MI455X (gfx1250) — matrix operand order repaired and checked by reading, not compiled, not run on hardware
//
#include <hip/hip_runtime.h>
#include <hip/hip_bf16.h>
#include <stdint.h>

// ---------- CDNA5 WMMA / TDM types ----------
typedef __attribute__((ext_vector_type(16))) __bf16 v16bf;
typedef __attribute__((ext_vector_type(8)))  float  v8f;
typedef unsigned int v4u __attribute__((ext_vector_type(4)));
typedef int v4i __attribute__((ext_vector_type(4)));
typedef int v8i __attribute__((ext_vector_type(8)));

union FragU { uint4 q[2]; v16bf v; };

__device__ __forceinline__ v8f wmma_bf16(v16bf a, v16bf b, v8f c) {
  // D = A(16x32) * B(32x16) + C  (f32 accum)
  return __builtin_amdgcn_wmma_f32_16x16x32_bf16(
      /*neg_a=*/false, a, /*neg_b=*/false, b,
      /*c_mod=*/(short)0, c, /*reuse_a=*/false, /*reuse_b=*/false);
}

// f32 -> bf16 round-to-nearest-even
__device__ __forceinline__ unsigned short f2bf(float x) {
  unsigned u = __float_as_uint(x);
  u += 0x7FFFu + ((u >> 16) & 1u);
  return (unsigned short)(u >> 16);
}

// A-fragment (16x32, M x K) from row-major [M][K]; rowptr already includes this
// lane's row (m = lane&15) and the k-tile base. Lane group g = lane>>4:
//   dwords 0-3: K = g*8 .. g*8+7 ; dwords 4-7: K = g*8+16 .. g*8+23
__device__ __forceinline__ v16bf load_a_frag(const unsigned short* rowptr, int g) {
  FragU f;
  const unsigned short* p = rowptr + g * 8;
  f.q[0] = *reinterpret_cast<const uint4*>(p);
  f.q[1] = *reinterpret_cast<const uint4*>(p + 16);
  return f.v;
}

// B-fragment (32x16, K x N) from row-major Bt[N][K]; colptr already includes
// this lane's column row (n = lane&15) and the k-tile base.
//   dwords 0-3: K = g*8 .. g*8+7 ; dwords 4-7: K = g*8+16 .. g*8+23
__device__ __forceinline__ v16bf load_b_frag(const unsigned short* colptr, int g) {
  FragU f;
  const unsigned short* p = colptr + g * 8;
  f.q[0] = *reinterpret_cast<const uint4*>(p);
  f.q[1] = *reinterpret_cast<const uint4*>(p + 16);
  return f.v;
}

#define NTOK   32768              // 8 frames * 4096 tokens
#define C      512
#define NSEQ   4096
#define ISC    0.044194173824159216f   // 512^-0.5
#define LOG2E  1.44269504088896340736f

// ---------- TDM: async-load a 32-row x 512-col bf16 tile into LDS ----------
// D# per CDNA5 ISA §8.3/8.4. 2-D tile: trailing descriptor groups zero-filled.
// This toolchain exposes the 6-arg builtin:
//   (uint32x4 g0, int32x8 g1, int32x4, int32x4, int32x8, i32 cpol)
__device__ __forceinline__ void tdm_load_k_tile(const unsigned short* gsrc,
                                                unsigned ldsDst) {
  unsigned long long ga = (unsigned long long)(uintptr_t)gsrc;
  v4u g0;
  g0[0] = 1u;                                    // count=1, user descriptor
  g0[1] = ldsDst;                                // lds_addr (bytes)
  g0[2] = (unsigned)(ga & 0xFFFFFFFFull);        // global_addr[31:0]
  g0[3] = (unsigned)(ga >> 32) | 0x80000000u;    // global_addr[56:32] | type=2
  v8i g1;
  g1[0] = 0x00010000;            // workgroup_mask=0, data_size=1 (2 bytes)
  g1[1] = (int)(512u << 16);     // tensor_dim0 = 512 (K elements per row)
  g1[2] = (int)(4096u << 16);    // tensor_dim0_hi=0 ; tensor_dim1 = 4096 rows
  g1[3] = (int)(512u << 16);     // tensor_dim1_hi=0 ; tile_dim0 = 512
  g1[4] = 32;                    // tile_dim1 = 32 rows ; tile_dim2 = 0 (2-D)
  g1[5] = 512;                   // tensor_dim0_stride = 512 elements
  g1[6] = 0;                     // stride0_hi / tensor_dim1_stride lo (unused)
  g1[7] = 0;
  v4i gz4 = {0, 0, 0, 0};
  v8i gz8 = {0, 0, 0, 0, 0, 0, 0, 0};
  __builtin_amdgcn_tensor_load_to_lds(g0, g1, gz4, gz4, gz8, 0);
}

// ---------- Kernel 0: weight transpose + f32->bf16 ----------
__global__ void prep_weights(const float* __restrict__ qkv_w,
                             const float* __restrict__ proj_w,
                             unsigned short* __restrict__ wTq,
                             unsigned short* __restrict__ wTp) {
  int idx = blockIdx.x * 256 + threadIdx.x;
  if (idx < 1536 * 512) {               // wTq[n][k] = qkv_w[k][n]
    int n = idx >> 9, kk = idx & 511;
    wTq[idx] = f2bf(qkv_w[(size_t)kk * 1536 + n]);
  }
  int idx2 = idx - 1536 * 512;
  if (idx2 >= 0 && idx2 < 512 * 512) {  // wTp[n][k] = proj_w[k][n]
    int n = idx2 >> 9, kk = idx2 & 511;
    wTp[idx2] = f2bf(proj_w[(size_t)kk * 512 + n]);
  }
}

// ---------- Kernel 1: RMSNorm -> bf16 ----------
__global__ void rmsnorm_kernel(const float* __restrict__ x,
                               const float* __restrict__ scale,
                               unsigned short* __restrict__ xn) {
  int tok = blockIdx.x;        // 32768 tokens
  int tid = threadIdx.x;       // 128 threads, 4 f32 each
  const float4* xp = reinterpret_cast<const float4*>(x + (size_t)tok * C);
  float4 v = xp[tid];
  float ss = v.x * v.x + v.y * v.y + v.z * v.z + v.w * v.w;
#pragma unroll
  for (int msk = 1; msk < 32; msk <<= 1) ss += __shfl_xor(ss, msk, 32);
  __shared__ float part[4];
  if ((tid & 31) == 0) part[tid >> 5] = ss;
  __syncthreads();
  float tot = part[0] + part[1] + part[2] + part[3];
  float f = rsqrtf(tot * (1.0f / 512.0f) + 1e-6f) * 22.627416997969522f; // *sqrt(c)
  float4 sc = reinterpret_cast<const float4*>(scale)[tid];
  uint2 o;
  o.x = (unsigned)f2bf(v.x * f * sc.x) | ((unsigned)f2bf(v.y * f * sc.y) << 16);
  o.y = (unsigned)f2bf(v.z * f * sc.z) | ((unsigned)f2bf(v.w * f * sc.w) << 16);
  *reinterpret_cast<uint2*>(xn + (size_t)tok * C + tid * 4) = o;
}

// ---------- Kernel 2: QKV GEMM  [32768,512] x [512,1536] ----------
__global__ void qkv_gemm(const unsigned short* __restrict__ xn,
                         const unsigned short* __restrict__ wTq,
                         const float* __restrict__ qkv_b,
                         unsigned short* __restrict__ qb,
                         unsigned short* __restrict__ kb,
                         unsigned short* __restrict__ vT) {
  int lane = threadIdx.x & 31, wv = threadIdx.x >> 5;
  int m = lane & 15, g = lane >> 4, nl = m;
  int rowTile = blockIdx.x * 4 + wv;   // 0..2047
  int n0 = blockIdx.y * 16;            // 0..1535
  const unsigned short* arow = xn + ((size_t)rowTile * 16 + m) * C;
  const unsigned short* brow = wTq + (size_t)(n0 + nl) * C;
  v8f acc = {};
#pragma unroll
  for (int kt = 0; kt < 16; ++kt) {
    v16bf a = load_a_frag(arow + kt * 32, g);
    v16bf b = load_b_frag(brow + kt * 32, g);
    acc = wmma_bf16(a, b, acc);
  }
  float bias = qkv_b[n0 + nl];
  int rowBase = rowTile * 16;
  if (n0 < 512) {                       // Q, pre-scaled by c^-0.5
#pragma unroll
    for (int r = 0; r < 8; ++r) {
      int tr = rowBase + r + 8 * g;
      qb[(size_t)tr * C + n0 + nl] = f2bf((acc[r] + bias) * ISC);
    }
  } else if (n0 < 1024) {               // K row-major
#pragma unroll
    for (int r = 0; r < 8; ++r) {
      int tr = rowBase + r + 8 * g;
      kb[(size_t)tr * C + (n0 - 512) + nl] = f2bf(acc[r] + bias);
    }
  } else {                              // V transposed: vT[frame][c][j]
    int c = n0 - 1024 + nl;
#pragma unroll
    for (int r = 0; r < 8; ++r) {
      int tr = rowBase + r + 8 * g;
      int frame = tr >> 12, ti = tr & 4095;
      vT[(size_t)frame * C * NSEQ + (size_t)c * NSEQ + ti] = f2bf(acc[r] + bias);
    }
  }
}

// ---------- Kernel 3: flash attention with TDM-staged K tiles ----------
// block = 2 waves; wave handles 16 query rows x 256 channels.
// K tiles (32x512 bf16 = 32KB) are DMA'd into a double-buffered LDS region by
// the Tensor Data Mover, overlapped with compute; both waves consume them via
// ds_load_b128 B-fragments (no duplicated global K traffic).
__global__ void __launch_bounds__(64, 1)
attn_kernel(const unsigned short* __restrict__ qb,
            const unsigned short* __restrict__ kb,
            const unsigned short* __restrict__ vT,
            unsigned short* __restrict__ ob) {
  int lane = threadIdx.x & 31, wv = threadIdx.x >> 5;
  int m = lane & 15, g = lane >> 4, nl = m;
  int row0 = blockIdx.x * 16;          // 0..4080
  int frame = blockIdx.y;              // 0..7
  size_t fbase = (size_t)frame * NSEQ;
  int c0 = wv * 256;

  __shared__ __align__(16) unsigned short ktile[2][32 * C];   // 64 KB
  __shared__ __align__(16) unsigned short ptile[2][16 * 32];  // 2 KB
  unsigned short* pt = &ptile[wv][0];

  // Q fragments held in registers for the whole KV sweep (16 x v16bf)
  const unsigned short* qrow = qb + (fbase + row0 + m) * C;
  v16bf qf[16];
#pragma unroll
  for (int kt = 0; kt < 16; ++kt) qf[kt] = load_a_frag(qrow + kt * 32, g);

  v8f zero = {};
  v8f acc[16];
#pragma unroll
  for (int t = 0; t < 16; ++t) acc[t] = zero;
  float mrow[8], lrow[8];
#pragma unroll
  for (int r = 0; r < 8; ++r) { mrow[r] = -3.0e38f; lrow[r] = 0.f; }

  // prologue: DMA first K tile
  if (wv == 0)
    tdm_load_k_tile(kb + fbase * C, (unsigned)(uintptr_t)&ktile[0][0]);

  for (int jt = 0; jt < 128; ++jt) {
    int j0 = jt * 32;
    __syncthreads();   // all waves done reading the buffer about to be refilled
    if (wv == 0) {
      if (jt + 1 < 128) {
        tdm_load_k_tile(kb + (fbase + j0 + 32) * C,
                        (unsigned)(uintptr_t)&ktile[(jt + 1) & 1][0]);
        __builtin_amdgcn_s_wait_tensorcnt((short)1);  // tile jt complete
      } else {
        __builtin_amdgcn_s_wait_tensorcnt((short)0);
      }
    }
    __syncthreads();   // tile jt visible to both waves

    // ---- S = Q K^T (16x32 tile = two 16x16 WMMA accumulators), K from LDS ----
    const unsigned short* ktl = &ktile[jt & 1][0];
    const unsigned short* k0p = ktl + (size_t)nl * C;
    const unsigned short* k1p = ktl + (size_t)(16 + nl) * C;
    v8f s0 = zero, s1 = zero;
#pragma unroll
    for (int kt = 0; kt < 16; ++kt) {
      v16bf b0 = load_b_frag(k0p + kt * 32, g);
      v16bf b1 = load_b_frag(k1p + kt * 32, g);
      s0 = wmma_bf16(qf[kt], b0, s0);
      s1 = wmma_bf16(qf[kt], b1, s1);
    }
    // ---- online softmax over the 32 keys of this step ----
    float corr[8];
#pragma unroll
    for (int r = 0; r < 8; ++r) {
      float v = fmaxf(s0[r], s1[r]);
      v = fmaxf(v, __shfl_xor(v, 1, 32));
      v = fmaxf(v, __shfl_xor(v, 2, 32));
      v = fmaxf(v, __shfl_xor(v, 4, 32));
      v = fmaxf(v, __shfl_xor(v, 8, 32));
      float nm = fmaxf(mrow[r], v);
      float cr = exp2f((mrow[r] - nm) * LOG2E);
      float p0 = exp2f((s0[r] - nm) * LOG2E);
      float p1 = exp2f((s1[r] - nm) * LOG2E);
      float rs = p0 + p1;
      rs += __shfl_xor(rs, 1, 32);
      rs += __shfl_xor(rs, 2, 32);
      rs += __shfl_xor(rs, 4, 32);
      rs += __shfl_xor(rs, 8, 32);
      lrow[r] = lrow[r] * cr + rs;
      mrow[r] = nm;
      corr[r] = cr;
      // stage P (bf16) in LDS in row-major [16][32] for A-fragment reload
      pt[(r + 8 * g) * 32 + nl] = f2bf(p0);
      pt[(r + 8 * g) * 32 + nl + 16] = f2bf(p1);
    }
    asm volatile("s_wait_dscnt 0x0" ::: "memory");
    v16bf pa = load_a_frag(pt + m * 32, g);
    // rescale running O by exp(m_old - m_new)
#pragma unroll
    for (int t = 0; t < 16; ++t) {
#pragma unroll
      for (int r = 0; r < 8; ++r) acc[t][r] *= corr[r];
    }
    // ---- O += P V  (V transposed: vT[c][j], B-fragment contiguous in j) ----
    const unsigned short* vcol =
        vT + (size_t)frame * C * NSEQ + (size_t)(c0 + nl) * NSEQ + j0;
#pragma unroll
    for (int t = 0; t < 16; ++t) {
      v16bf vb = load_b_frag(vcol + (size_t)t * 16 * NSEQ, g);
      acc[t] = wmma_bf16(pa, vb, acc[t]);
    }
  }
  // ---- epilogue: O / l -> bf16 ----
#pragma unroll
  for (int r = 0; r < 8; ++r) {
    float inv = 1.0f / lrow[r];
    size_t orow = (fbase + row0 + r + 8 * g) * C;
#pragma unroll
    for (int t = 0; t < 16; ++t)
      ob[orow + c0 + t * 16 + nl] = f2bf(acc[t][r] * inv);
  }
}

// ---------- Kernel 4: proj GEMM + bias + residual ----------
__global__ void proj_gemm(const unsigned short* __restrict__ ob,
                          const unsigned short* __restrict__ wTp,
                          const float* __restrict__ proj_b,
                          const float* __restrict__ x,
                          float* __restrict__ out) {
  int lane = threadIdx.x & 31, wv = threadIdx.x >> 5;
  int m = lane & 15, g = lane >> 4, nl = m;
  int rowTile = blockIdx.x * 4 + wv;   // 0..2047
  int n0 = blockIdx.y * 16;            // 0..511
  const unsigned short* arow = ob + ((size_t)rowTile * 16 + m) * C;
  const unsigned short* brow = wTp + (size_t)(n0 + nl) * C;
  v8f acc = {};
#pragma unroll
  for (int kt = 0; kt < 16; ++kt) {
    v16bf a = load_a_frag(arow + kt * 32, g);
    v16bf b = load_b_frag(brow + kt * 32, g);
    acc = wmma_bf16(a, b, acc);
  }
  float bias = proj_b[n0 + nl];
#pragma unroll
  for (int r = 0; r < 8; ++r) {
    size_t row = (size_t)rowTile * 16 + r + 8 * g;
    size_t idx = row * C + n0 + nl;
    out[idx] = acc[r] + bias + x[idx];
  }
}

// ---------- launch ----------
extern "C" void kernel_launch(void* const* d_in, const int* in_sizes, int n_in,
                              void* d_out, int out_size, void* d_ws, size_t ws_size,
                              hipStream_t stream) {
  const float* x      = (const float*)d_in[0];
  const float* scale  = (const float*)d_in[1];
  const float* qkv_w  = (const float*)d_in[2];
  const float* qkv_b  = (const float*)d_in[3];
  const float* proj_w = (const float*)d_in[4];
  const float* proj_b = (const float*)d_in[5];
  float* out = (float*)d_out;

  const size_t SZ = (size_t)NTOK * C * 2;   // 32 MB per bf16 activation buffer
  char* ws = (char*)d_ws;
  unsigned short* qb  = (unsigned short*)(ws);
  unsigned short* kb  = (unsigned short*)(ws + SZ);
  unsigned short* vT  = (unsigned short*)(ws + 2 * SZ);
  unsigned short* xn  = (unsigned short*)(ws + 3 * SZ);  // reused as attn output
  unsigned short* wTq = (unsigned short*)(ws + 4 * SZ);
  unsigned short* wTp = (unsigned short*)(ws + 4 * SZ + (size_t)1536 * 512 * 2);

  prep_weights<<<4096, 256, 0, stream>>>(qkv_w, proj_w, wTq, wTp);
  rmsnorm_kernel<<<NTOK, 128, 0, stream>>>(x, scale, xn);
  qkv_gemm<<<dim3(512, 96), 128, 0, stream>>>(xn, wTq, qkv_b, qb, kb, vT);
  attn_kernel<<<dim3(256, 8), 64, 0, stream>>>(qb, kb, vT, xn);
  proj_gemm<<<dim3(512, 32), 128, 0, stream>>>(xn, wTp, proj_b, x, out);
}
